// Model_16612933501124
// MI455X (gfx1250) — matrix operand order repaired and checked by reading, not compiled, not run on hardware
//
#include <hip/hip_runtime.h>

typedef __attribute__((ext_vector_type(16))) _Float16 v16h;
typedef __attribute__((ext_vector_type(8)))  _Float16 v8h;
typedef __attribute__((ext_vector_type(8)))  float    v8f;

namespace {
constexpr int cB = 8, cT = 12, cN = 716, cH = 128, cDEG = 12, cL = 5;
constexpr int cKS[cL]  = {11, 9, 7, 4, 1};
constexpr int cKIN[cL] = {12, 11, 9, 7, 4};
constexpr int cDIL[cL] = {1, 2, 2, 3, 3};
// LDS weight tile layout: row stride padded to 136 halfs to spread banks
constexpr int cLWR = 136;                 // halfs per row in LDS
constexpr int cLWM = cH * cLWR;           // halfs per matrix in LDS (17408)
constexpr size_t cSMEM = (size_t)5 * cLWM * 2 + 2 * 256 * 4;  // 5 matrices + reduction
}

// ---------------- input projection: last = inputs @ W_in + b_in ----------------
__global__ void k_inproj(const float* __restrict__ inp, const float* __restrict__ Win,
                         const float* __restrict__ bin, float* __restrict__ last, int total) {
  int i = blockIdx.x * blockDim.x + threadIdx.x;
  if (i >= total) return;
  int h   = i & (cH - 1);
  int btn = i >> 7;
  float x0 = inp[btn * 2 + 0];
  float x1 = inp[btn * 2 + 1];
  last[i] = fmaf(x0, Win[h], fmaf(x1, Win[cH + h], bin[h]));
}

// ------------- convert one 128x128 fp32 weight to f16, transposed (n-major) -------------
__global__ void k_wcvt(const float* __restrict__ W, _Float16* __restrict__ WT) {
  int i = blockIdx.x * blockDim.x + threadIdx.x;   // 0..16383
  int k = i >> 7, n = i & 127;
  WT[n * 128 + k] = (_Float16)W[k * 128 + n];
}

// ------------- temporal mixes: residual (fp32) and gate pre-mix gf0 (f16) -------------
template <int KIN, int K>
__global__ void k_temporal(const float* __restrict__ last,
                           const float* __restrict__ Wres, const float* __restrict__ bres,
                           const float* __restrict__ Wgate, const float* __restrict__ bgate,
                           float* __restrict__ resid, _Float16* __restrict__ gf0h) {
  int i = blockIdx.x * blockDim.x + threadIdx.x;      // (b, n, h)
  if (i >= cB * cN * cH) return;
  int h = i & 127;
  int bn = i >> 7;
  int b = bn / cN, n = bn - b * cN;
  float x[KIN];
#pragma unroll
  for (int t = 0; t < KIN; ++t)
    x[t] = last[((size_t)(b * KIN + t) * cN + n) * cH + h];
#pragma unroll
  for (int k = 0; k < K; ++k) {
    float r = 0.f, g = 0.f;
#pragma unroll
    for (int t = 0; t < KIN; ++t) {
      r = fmaf(x[t], Wres[k * KIN + t], r);
      g = fmaf(x[t], Wgate[k * KIN + t], g);
    }
    size_t o = ((size_t)(b * K + k) * cN + n) * cH + h;
    resid[o] = r + bres[k];
    gf0h[o]  = (_Float16)(g + bgate[k]);
  }
}

// ------------- graph gather: agg = seg-sum(ew * hl[src]);  fwd = agg + hr -------------
__global__ void k_gather(const float* __restrict__ last, const int* __restrict__ src,
                         const float* __restrict__ ew,
                         _Float16* __restrict__ aggh, _Float16* __restrict__ fwdh,
                         int kin, int K, int dil, int total) {
  int i = blockIdx.x * blockDim.x + threadIdx.x;   // (b,k,n,h)
  if (i >= total) return;
  int h = i & 127;
  int bkn = i >> 7;
  int n = bkn % cN;
  int bk = bkn / cN;
  int k = bk % K;
  int b = bk / K;
  size_t rowbase = (size_t)(b * kin + k) * cN;
  float a = 0.f;
  int e = n * cDEG;
#pragma unroll
  for (int j = 0; j < cDEG; ++j) {
    int s = src[e + j];
    a = fmaf(ew[e + j], last[(rowbase + s) * cH + h], a);
  }
  float hr = last[((size_t)(b * kin + k + dil) * cN + n) * cH + h];
  aggh[i] = (_Float16)a;
  fwdh[i] = (_Float16)(a + hr);
}

__global__ void k_zero2(float* s) { s[0] = 0.f; s[1] = 0.f; }

// ---------------- WMMA fragment helpers (layouts per CDNA5 ISA) ----------------
__device__ __forceinline__ v16h make16(v8h lo, v8h hi) {
  v16h r;
#pragma unroll
  for (int j = 0; j < 8; ++j) { r[j] = lo[j]; r[8 + j] = hi[j]; }
  return r;
}
// A 16x32 f16: lane row m = lane&15; e<8 -> k = kt*32 + hf*8 + e ; e>=8 -> k = kt*32 + 16 + hf*8 + (e-8)
__device__ __forceinline__ v16h load_afrag(const _Float16* __restrict__ row, int kt, int hf) {
  v8h lo = *(const v8h*)(row + kt * 32 + hf * 8);
  v8h hi = *(const v8h*)(row + kt * 32 + 16 + hf * 8);
  return make16(lo, hi);
}
// B 32x16 f16 from LDS (n-major, padded rows): lane col n = lane&15; e<8 -> k = kt*32 + hf*8 + e ; e>=8 -> k = kt*32 + 16 + hf*8 + (e-8)
__device__ __forceinline__ v16h load_bfrag_lds(const _Float16* mat, int n, int kt, int hf) {
  const _Float16* p = mat + n * cLWR + kt * 32 + hf * 8;
  v8h lo = *(const v8h*)(p);
  v8h hi = *(const v8h*)(p + 16);
  return make16(lo, hi);
}

// ------- fused: stage 5 weight matrices in LDS, 5 WMMA GEMMs + gate combine + LN sums -------
// conv buffer holds `residual` on entry, `conv_res` on exit.
__global__ void __launch_bounds__(256)
k_fused(const _Float16* __restrict__ gf0h, const _Float16* __restrict__ aggh,
        const _Float16* __restrict__ fwdh,
        const _Float16* __restrict__ Wg1t, const _Float16* __restrict__ Wa1t,
        const _Float16* __restrict__ Wa2t, const _Float16* __restrict__ Wp1t,
        const _Float16* __restrict__ Wp2t,
        const float* __restrict__ bg1, const float* __restrict__ ba,
        const float* __restrict__ bp,
        float* __restrict__ conv, float* __restrict__ sums, int M) {
  extern __shared__ char smem[];
  _Float16* lw  = (_Float16*)smem;                       // 5 * cLWM halfs
  float*    red = (float*)(smem + (size_t)5 * cLWM * 2); // 2 * 256 floats

  int tid  = threadIdx.x;
  int wave = tid >> 5;
  int lane = tid & 31;
  int hf = lane >> 4;
  int lh = lane & 15;
  int row0 = (blockIdx.x * 8 + wave) * 16;
  bool active = row0 < M;                        // wave-uniform

  // ---- issue A-fragment global loads first (overlap with LDS staging) ----
  v16h Ag[4], Aa[4], Af[4];
  if (active) {
    int ra = row0 + lh; if (ra >= M) ra = M - 1; // clamp loads; stores masked below
    const _Float16* grow = gf0h + (size_t)ra * cH;
    const _Float16* arow = aggh + (size_t)ra * cH;
    const _Float16* frow = fwdh + (size_t)ra * cH;
#pragma unroll
    for (int kt = 0; kt < 4; ++kt) {
      Ag[kt] = load_afrag(grow, kt, hf);
      Aa[kt] = load_afrag(arow, kt, hf);
      Af[kt] = load_afrag(frow, kt, hf);
    }
  }

  // ---- cooperative staging: 5 matrices, 128 rows x 16 chunks of 8 halfs each ----
  {
    const _Float16* srcs[5] = {Wg1t, Wa1t, Wa2t, Wp1t, Wp2t};
#pragma unroll
    for (int m = 0; m < 5; ++m) {
      const _Float16* s = srcs[m];
      _Float16* d = lw + m * cLWM;
      for (int c = tid; c < 128 * 16; c += 256) {   // 8 iters per thread
        int n = c >> 4, part = c & 15;
        *(v8h*)(d + n * cLWR + part * 8) = *(const v8h*)(s + n * 128 + part * 8);
      }
    }
  }
  __syncthreads();

  float lsum = 0.f, lsq = 0.f;
  if (active) {                                  // wave-uniform: EXEC all-ones for WMMA
    const _Float16* Lg1 = lw + 0 * cLWM;
    const _Float16* La1 = lw + 1 * cLWM;
    const _Float16* La2 = lw + 2 * cLWM;
    const _Float16* Lp1 = lw + 3 * cLWM;
    const _Float16* Lp2 = lw + 4 * cLWM;
    for (int nt = 0; nt < 8; ++nt) {             // 8 column tiles of 16 -> H=128
      int nb = nt * 16 + lh;
      v8f accg = {}, acca = {}, accp = {};
#pragma unroll
      for (int kt = 0; kt < 4; ++kt) {
        v16h b0 = load_bfrag_lds(Lg1, nb, kt, hf);
        accg = __builtin_amdgcn_wmma_f32_16x16x32_f16(false, Ag[kt], false, b0, (short)0, accg, false, false);
        v16h b1 = load_bfrag_lds(La1, nb, kt, hf);
        acca = __builtin_amdgcn_wmma_f32_16x16x32_f16(false, Aa[kt], false, b1, (short)0, acca, false, false);
        v16h b2 = load_bfrag_lds(La2, nb, kt, hf);
        acca = __builtin_amdgcn_wmma_f32_16x16x32_f16(false, Af[kt], false, b2, (short)0, acca, false, false);
        v16h b3 = load_bfrag_lds(Lp1, nb, kt, hf);
        accp = __builtin_amdgcn_wmma_f32_16x16x32_f16(false, Aa[kt], false, b3, (short)0, accp, false, false);
        v16h b4 = load_bfrag_lds(Lp2, nb, kt, hf);
        accp = __builtin_amdgcn_wmma_f32_16x16x32_f16(false, Af[kt], false, b4, (short)0, accp, false, false);
      }
      float biasg = bg1[nb], biasa = ba[nb], biasp = bp[nb];
#pragma unroll
      for (int r = 0; r < 8; ++r) {              // C/D layout: m = row0 + hf*8 + r, n = nb
        int m = row0 + hf * 8 + r;
        float gfv  = fmaxf(accg[r] + biasg, 0.f);
        float gate = 1.f / (1.f + __expf(-gfv));
        float adj  = fmaxf(acca[r] + biasa, 0.f);
        float pea  = fmaxf(accp[r] + biasp, 0.f);
        if (m < M) {
          size_t o = (size_t)m * cH + nb;
          float cv = fmaf(gfv, adj, fmaf(1.f - gate, pea, conv[o]));
          conv[o] = cv;
          lsum += cv;
          lsq = fmaf(cv, cv, lsq);
        }
      }
    }
  }
  // ---- block reduction of layernorm partial sums ----
  red[tid] = lsum; red[256 + tid] = lsq;
  __syncthreads();
  for (int s = 128; s > 0; s >>= 1) {
    if (tid < s) { red[tid] += red[tid + s]; red[256 + tid] += red[256 + tid + s]; }
    __syncthreads();
  }
  if (tid == 0) { atomicAdd(&sums[0], red[0]); atomicAdd(&sums[1], red[256]); }
}

// ------------- full-tensor layernorm finalize; writes next `last` in place -------------
__global__ void k_lnorm(const float* __restrict__ conv, const float* __restrict__ gamma,
                        const float* __restrict__ beta, const float* __restrict__ sums,
                        float* __restrict__ lastOut, int total) {
  int i = blockIdx.x * blockDim.x + threadIdx.x;
  if (i >= total) return;
  float cnt = (float)total;
  float mu  = sums[0] / cnt;
  float var = sums[1] / cnt - mu * mu;
  float rs  = rsqrtf(var + 1e-5f);
  lastOut[i] = fmaf(gamma[i], (conv[i] - mu) * rs, beta[i]);
}

// ------------- decode: relu(last * Wdec[p] + bdec[p]) . Wdl  + bdl -------------
__global__ void k_decode(const float* __restrict__ last, const float* __restrict__ Wdec,
                         const float* __restrict__ bdec, const float* __restrict__ Wdl,
                         const float* __restrict__ bdl, float* __restrict__ out) {
  int bn = blockIdx.x;                 // (b, n), last is (B,1,N,H)
  int b = bn / cN, n = bn - b * cN;
  int h = threadIdx.x;                 // 128 threads
  float x  = last[(size_t)bn * cH + h];
  float wl = Wdl[h];
  __shared__ float sd[128];
  for (int p = 0; p < 12; ++p) {
    float v = fmaxf(fmaf(x, Wdec[p], bdec[p]), 0.f) * wl;
    sd[h] = v; __syncthreads();
    for (int s = 64; s > 0; s >>= 1) { if (h < s) sd[h] += sd[h + s]; __syncthreads(); }
    if (h == 0) out[(size_t)(b * 12 + p) * cN + n] = sd[0] + bdl[0];
    __syncthreads();
  }
}

extern "C" void kernel_launch(void* const* d_in, const int* in_sizes, int n_in,
                              void* d_out, int out_size, void* d_ws, size_t ws_size,
                              hipStream_t stream) {
  (void)in_sizes; (void)n_in; (void)out_size; (void)ws_size;
  // dict order with lists flattened elementwise:
  const float* inputs = (const float*)d_in[0];
  const float* W_in = (const float*)d_in[1];
  const float* b_in = (const float*)d_in[2];
  const float* Wres[cL];  for (int i = 0; i < cL; ++i) Wres[i]  = (const float*)d_in[3 + i];
  const float* bres[cL];  for (int i = 0; i < cL; ++i) bres[i]  = (const float*)d_in[8 + i];
  const float* Wgate[cL]; for (int i = 0; i < cL; ++i) Wgate[i] = (const float*)d_in[13 + i];
  const float* bgate[cL]; for (int i = 0; i < cL; ++i) bgate[i] = (const float*)d_in[18 + i];
  const float* Wg1 = (const float*)d_in[23];
  const float* bg1 = (const float*)d_in[24];
  const float* Wa1 = (const float*)d_in[25];     // (L,H,H)
  const float* Wa2 = (const float*)d_in[26];
  const float* ba  = (const float*)d_in[27];     // (L,H)
  const float* Wp1 = (const float*)d_in[28];
  const float* Wp2 = (const float*)d_in[29];
  const float* bp  = (const float*)d_in[30];
  const float* gammas[cL]; for (int i = 0; i < cL; ++i) gammas[i] = (const float*)d_in[31 + i];
  const float* betas[cL];  for (int i = 0; i < cL; ++i) betas[i]  = (const float*)d_in[36 + i];
  const float* Wdec = (const float*)d_in[41];
  const float* bdec = (const float*)d_in[42];
  const float* Wdl  = (const float*)d_in[43];
  const float* bdl  = (const float*)d_in[44];
  const float* ew   = (const float*)d_in[45];
  const int*   src  = (const int*)d_in[46];
  // d_in[47] = dst (implied by repeat(arange(N), DEG)); unused.

  char* wsp = (char*)d_ws;
  auto carve = [&](size_t bytes) { char* p = wsp; wsp += (bytes + 255) & ~(size_t)255; return p; };
  const size_t actMax = (size_t)cB * 11 * cN * cH;
  float*    lastF = (float*)carve((size_t)cB * cT * cN * cH * 4);  // ping buffer (T=12 slots)
  float*    resF  = (float*)carve(actMax * 4);                     // residual -> conv
  _Float16* gf0h  = (_Float16*)carve(actMax * 2);
  _Float16* aggh  = (_Float16*)carve(actMax * 2);
  _Float16* fwdh  = (_Float16*)carve(actMax * 2);
  _Float16* wts   = (_Float16*)carve((size_t)21 * cH * cH * 2);    // [Wg1 | Wa1 x5 | Wa2 x5 | Wp1 x5 | Wp2 x5]
  float*    sums  = (float*)carve(256);

  // f16 + transpose all H x H weights once
  k_wcvt<<<64, 256, 0, stream>>>(Wg1, wts);
  for (int i = 0; i < cL; ++i) {
    k_wcvt<<<64, 256, 0, stream>>>(Wa1 + (size_t)i * cH * cH, wts + (size_t)(1 + i)  * cH * cH);
    k_wcvt<<<64, 256, 0, stream>>>(Wa2 + (size_t)i * cH * cH, wts + (size_t)(6 + i)  * cH * cH);
    k_wcvt<<<64, 256, 0, stream>>>(Wp1 + (size_t)i * cH * cH, wts + (size_t)(11 + i) * cH * cH);
    k_wcvt<<<64, 256, 0, stream>>>(Wp2 + (size_t)i * cH * cH, wts + (size_t)(16 + i) * cH * cH);
  }

  int tot0 = cB * cT * cN * cH;
  k_inproj<<<(tot0 + 255) / 256, 256, 0, stream>>>(inputs, W_in, b_in, lastF, tot0);

  int bnTot = cB * cN * cH;
  for (int i = 0; i < cL; ++i) {
    int K = cKS[i], kin = cKIN[i], dil = cDIL[i];
    int M = cB * K * cN;
    int tot = M * cH;
    dim3 gT((bnTot + 255) / 256), bl(256);
    switch (i) {
      case 0: k_temporal<12, 11><<<gT, bl, 0, stream>>>(lastF, Wres[i], bres[i], Wgate[i], bgate[i], resF, gf0h); break;
      case 1: k_temporal<11,  9><<<gT, bl, 0, stream>>>(lastF, Wres[i], bres[i], Wgate[i], bgate[i], resF, gf0h); break;
      case 2: k_temporal< 9,  7><<<gT, bl, 0, stream>>>(lastF, Wres[i], bres[i], Wgate[i], bgate[i], resF, gf0h); break;
      case 3: k_temporal< 7,  4><<<gT, bl, 0, stream>>>(lastF, Wres[i], bres[i], Wgate[i], bgate[i], resF, gf0h); break;
      case 4: k_temporal< 4,  1><<<gT, bl, 0, stream>>>(lastF, Wres[i], bres[i], Wgate[i], bgate[i], resF, gf0h); break;
    }
    k_gather<<<(tot + 255) / 256, 256, 0, stream>>>(lastF, src, ew, aggh, fwdh, kin, K, dil, tot);
    k_zero2<<<1, 1, 0, stream>>>(sums);
    int nblk = (M + 127) / 128;   // 8 waves/block, 16 rows/wave
    k_fused<<<nblk, 256, cSMEM, stream>>>(gf0h, aggh, fwdh,
        wts, wts + (size_t)(1 + i) * cH * cH, wts + (size_t)(6 + i) * cH * cH,
        wts + (size_t)(11 + i) * cH * cH, wts + (size_t)(16 + i) * cH * cH,
        bg1, ba + (size_t)i * cH, bp + (size_t)i * cH, resF, sums, M);
    k_lnorm<<<(tot + 255) / 256, 256, 0, stream>>>(resF, gammas[i], betas[i], sums, lastF, tot);
  }
  k_decode<<<cB * cN, 128, 0, stream>>>(lastF, Wdec, bdec, Wdl, bdl, (float*)d_out);
}
